// EpiPINN_34797825032888
// MI455X (gfx1250) — matrix operand order repaired and checked by reading, not compiled, not run on hardware
//
#include <hip/hip_runtime.h>

#define NPTS      8192
#define HDIM      512
#define NLAYER    5
#define DNM1      8191        // NPTS - 1
#define DTSTEP    0.1f
#define MIN_ALPHA 0.6f

#define BM 128
#define BN 64
#define KT 32
#define NKT (HDIM / KT)

typedef __attribute__((ext_vector_type(16))) __bf16 v16bf;
typedef __attribute__((ext_vector_type(8)))  float  v8f;

union Frag { v16bf bf; unsigned int u[8]; };

__device__ __forceinline__ unsigned short f32_to_bf16(float f) {
  unsigned int u = __float_as_uint(f);
  u += 0x7fffu + ((u >> 16) & 1u);          // round-to-nearest-even
  return (unsigned short)(u >> 16);
}
__device__ __forceinline__ float bf16_to_f32(unsigned short h) {
  return __uint_as_float(((unsigned int)h) << 16);
}

// CDNA5 async global->LDS copy (bypasses VGPRs, tracked by ASYNCcnt)
__device__ __forceinline__ void async_cp16(unsigned lds_off, unsigned long long gaddr) {
  asm volatile("global_load_async_to_lds_b128 %0, %1, off"
               :: "v"(lds_off), "v"(gaddr) : "memory");
}
__device__ __forceinline__ void wait_async0() {
#if __has_builtin(__builtin_amdgcn_s_wait_asynccnt)
  __builtin_amdgcn_s_wait_asynccnt(0);
#else
  asm volatile("s_wait_asynccnt 0" ::: "memory");
#endif
}

// ---------------------------------------------------------------------------
// Split fp32 weights into hi/lo bf16 pair AND transpose to [l][n][k] so the
// GEMM's B tiles are contiguous 64B runs (async-copy friendly).
// ---------------------------------------------------------------------------
__global__ void k_split_w(const float* __restrict__ W,
                          unsigned short* __restrict__ hi,
                          unsigned short* __restrict__ lo) {
  const int idx = blockIdx.x * blockDim.x + threadIdx.x;   // l*H*H + k*H + n
  const int l = idx >> 18;                                  // H*H == 1<<18
  const int k = (idx >> 9) & (HDIM - 1);
  const int n = idx & (HDIM - 1);
  const float v = W[idx];
  const unsigned short h = f32_to_bf16(v);
  const int o = (l << 18) + (n << 9) + k;                   // transposed
  hi[o] = h;
  lo[o] = f32_to_bf16(v - bf16_to_f32(h));
}

// ---------------------------------------------------------------------------
// Input layer: h0 = tanh(t * W_in + b_in), emit hi/lo bf16 activations
// ---------------------------------------------------------------------------
__global__ void k_input_layer(const float* __restrict__ t,
                              const float* __restrict__ W_in,
                              const float* __restrict__ b_in,
                              unsigned short* __restrict__ Ohi,
                              unsigned short* __restrict__ Olo) {
  const int idx = blockIdx.x * blockDim.x + threadIdx.x;   // over NPTS*HDIM
  const int i = idx >> 9;
  const int j = idx & 511;
  const float v = tanhf(t[i] * W_in[j] + b_in[j]);
  const unsigned short h = f32_to_bf16(v);
  Ohi[idx] = h;
  Olo[idx] = f32_to_bf16(v - bf16_to_f32(h));
}

// ---------------------------------------------------------------------------
// Issue one K-tile's worth of async copies: A (128x32, hi+lo) and Bt (64x32,
// hi+lo). 256 threads * 6 x 16B lane-copies = 24 KB per tile.
// ---------------------------------------------------------------------------
__device__ __forceinline__ void stage_tile(
    const unsigned short* Ahi, const unsigned short* Alo,
    const unsigned short* Whi, const unsigned short* Wlo,
    int blockM, int blockN, int k0, int tid,
    unsigned aHi, unsigned aLo, unsigned bHi, unsigned bLo)
{
  const int r  = tid >> 2;              // 0..63
  const int sg = (tid & 3) * 16;        // 16B segment within a 64B row
  const size_t rowA0 = ((size_t)(blockM + r)      * HDIM + k0) * 2 + sg;
  const size_t rowA1 = ((size_t)(blockM + r + 64) * HDIM + k0) * 2 + sg;
  const size_t rowB  = ((size_t)(blockN + r)      * HDIM + k0) * 2 + sg;
  async_cp16(aHi + (unsigned)(r * 64 + sg),        (unsigned long long)Ahi + rowA0);
  async_cp16(aHi + (unsigned)((r + 64) * 64 + sg), (unsigned long long)Ahi + rowA1);
  async_cp16(aLo + (unsigned)(r * 64 + sg),        (unsigned long long)Alo + rowA0);
  async_cp16(aLo + (unsigned)((r + 64) * 64 + sg), (unsigned long long)Alo + rowA1);
  async_cp16(bHi + (unsigned)(r * 64 + sg),        (unsigned long long)Whi + rowB);
  async_cp16(bLo + (unsigned)(r * 64 + sg),        (unsigned long long)Wlo + rowB);
}

// ---------------------------------------------------------------------------
// Hidden layer: Hout = tanh(A @ W + b), split-bf16 WMMA (hi*hi + hi*lo + lo*hi)
// 256 threads = 8 waves; tile 128x64; wave tile 32x32; double-buffered LDS
// filled by async global->LDS copies overlapped with WMMA compute.
// ---------------------------------------------------------------------------
__global__ __launch_bounds__(256) void k_gemm_tanh(
    const unsigned short* __restrict__ Ahi, const unsigned short* __restrict__ Alo,
    const unsigned short* __restrict__ Whi, const unsigned short* __restrict__ Wlo,
    const float* __restrict__ bias,
    float* __restrict__ Hout,
    unsigned short* __restrict__ Ohi, unsigned short* __restrict__ Olo)
{
  __shared__ unsigned int sAhi[2][BM * KT / 2];   // 2 x 8 KB
  __shared__ unsigned int sAlo[2][BM * KT / 2];
  __shared__ unsigned int sBhi[2][BN * KT / 2];   // 2 x 4 KB, [n][k]
  __shared__ unsigned int sBlo[2][BN * KT / 2];

  const int tid  = threadIdx.x;
  const int lane = tid & 31;
  const int wave = tid >> 5;
  const int wm = wave >> 1;              // 0..3  (M direction)
  const int wn = wave & 1;               // 0..1  (N direction)
  const int blockM = blockIdx.x * BM;
  const int blockN = blockIdx.y * BN;
  const int nl   = lane & 15;
  const int half = lane >> 4;

  const unsigned aHiB[2] = {(unsigned)(unsigned long long)(void*)&sAhi[0][0],
                            (unsigned)(unsigned long long)(void*)&sAhi[1][0]};
  const unsigned aLoB[2] = {(unsigned)(unsigned long long)(void*)&sAlo[0][0],
                            (unsigned)(unsigned long long)(void*)&sAlo[1][0]};
  const unsigned bHiB[2] = {(unsigned)(unsigned long long)(void*)&sBhi[0][0],
                            (unsigned)(unsigned long long)(void*)&sBhi[1][0]};
  const unsigned bLoB[2] = {(unsigned)(unsigned long long)(void*)&sBlo[0][0],
                            (unsigned)(unsigned long long)(void*)&sBlo[1][0]};

  v8f acc[2][2];
  #pragma unroll
  for (int mi = 0; mi < 2; ++mi)
    #pragma unroll
    for (int ni = 0; ni < 2; ++ni)
      #pragma unroll
      for (int r = 0; r < 8; ++r) acc[mi][ni][r] = 0.0f;

  // prologue: fill buffer 0
  stage_tile(Ahi, Alo, Whi, Wlo, blockM, blockN, 0, tid,
             aHiB[0], aLoB[0], bHiB[0], bLoB[0]);
  wait_async0();
  __syncthreads();

  int cur = 0;
  for (int kt = 0; kt < NKT; ++kt) {
    // prefetch next tile into the other buffer while we compute
    if (kt + 1 < NKT)
      stage_tile(Ahi, Alo, Whi, Wlo, blockM, blockN, (kt + 1) * KT, tid,
                 aHiB[cur ^ 1], aLoB[cur ^ 1], bHiB[cur ^ 1], bLoB[cur ^ 1]);

    // B fragments: lane L holds N = L&15, dword d -> K = (d>=4?16:0)+8*half+(d&3)*2
    Frag bhf[2], blf[2];
    #pragma unroll
    for (int ni = 0; ni < 2; ++ni) {
      const int nn = wn * 32 + ni * 16 + nl;
      const unsigned int* ph = &sBhi[cur][0] + nn * (KT / 2) + half * 4;
      const unsigned int* pl = &sBlo[cur][0] + nn * (KT / 2) + half * 4;
      #pragma unroll
      for (int d = 0; d < 8; ++d) { bhf[ni].u[d] = ph[(d & 3) + ((d & 4) << 1)]; blf[ni].u[d] = pl[(d & 3) + ((d & 4) << 1)]; }
    }
    #pragma unroll
    for (int mi = 0; mi < 2; ++mi) {
      // A fragment (ISA 16-bit A layout): dword d -> K = (d>=4?16:0)+8*half+(d&3)*2
      Frag ah, al;
      const int mm = wm * 32 + mi * 16 + nl;
      #pragma unroll
      for (int d = 0; d < 8; ++d) {
        const int kb = ((d & 4) << 2) + half * 8 + (d & 3) * 2;
        ah.u[d] = sAhi[cur][mm * (KT / 2) + (kb >> 1)];
        al.u[d] = sAlo[cur][mm * (KT / 2) + (kb >> 1)];
      }
      #pragma unroll
      for (int ni = 0; ni < 2; ++ni) {
        acc[mi][ni] = __builtin_amdgcn_wmma_f32_16x16x32_bf16(
            false, ah.bf, false, bhf[ni].bf, (short)0, acc[mi][ni], false, false);
        acc[mi][ni] = __builtin_amdgcn_wmma_f32_16x16x32_bf16(
            false, ah.bf, false, blf[ni].bf, (short)0, acc[mi][ni], false, false);
        acc[mi][ni] = __builtin_amdgcn_wmma_f32_16x16x32_bf16(
            false, al.bf, false, bhf[ni].bf, (short)0, acc[mi][ni], false, false);
      }
    }
    wait_async0();      // own async copies into next buffer have landed in LDS
    __syncthreads();    // everyone done reading cur + everyone's copies visible
    cur ^= 1;
  }

  // Epilogue: C/D layout — VGPR r: M = r + 8*half, N = lane&15
  #pragma unroll
  for (int mi = 0; mi < 2; ++mi)
    #pragma unroll
    for (int ni = 0; ni < 2; ++ni) {
      const int col = blockN + wn * 32 + ni * 16 + nl;
      const float b = bias[col];
      #pragma unroll
      for (int r = 0; r < 8; ++r) {
        const int row = blockM + wm * 32 + mi * 16 + half * 8 + r;
        const float v = tanhf(acc[mi][ni][r] + b);
        const int o = row * HDIM + col;
        Hout[o] = v;
        const unsigned short hb = f32_to_bf16(v);
        Ohi[o] = hb;
        Olo[o] = f32_to_bf16(v - bf16_to_f32(hb));
      }
    }
}

// ---------------------------------------------------------------------------
// Output layer + softmax: one wave per row; [512]x[512,5] dot, wave32 reduce
// ---------------------------------------------------------------------------
__global__ __launch_bounds__(256) void k_out_softmax(
    const float* __restrict__ Hf, const float* __restrict__ Wout,
    const float* __restrict__ bout, float* __restrict__ y)
{
  const int lane = threadIdx.x & 31;
  const int row  = blockIdx.x * 8 + (threadIdx.x >> 5);
  float a[5] = {0.f, 0.f, 0.f, 0.f, 0.f};
  for (int j = lane; j < HDIM; j += 32) {
    const float h = Hf[row * HDIM + j];
    const float* w = Wout + j * 5;
    #pragma unroll
    for (int c = 0; c < 5; ++c) a[c] += h * w[c];
  }
  #pragma unroll
  for (int c = 0; c < 5; ++c)
    #pragma unroll
    for (int o = 16; o > 0; o >>= 1) a[c] += __shfl_xor(a[c], o, 32);
  #pragma unroll
  for (int c = 0; c < 5; ++c) a[c] += bout[c];
  float m = a[0];
  #pragma unroll
  for (int c = 1; c < 5; ++c) m = fmaxf(m, a[c]);
  float e[5]; float s = 0.f;
  #pragma unroll
  for (int c = 0; c < 5; ++c) { e[c] = expf(a[c] - m); s += e[c]; }
  if (lane == 0) {
    #pragma unroll
    for (int c = 0; c < 5; ++c) y[row * 5 + c] = e[c] / s;
  }
}

// ---------------------------------------------------------------------------
// Scalars (softplus params, alpha, C), L1 weights ws[r], dpsi (transposed)
// ---------------------------------------------------------------------------
__global__ void k_prep(const float* __restrict__ y,
                       const float* rb, const float* rs, const float* rg,
                       const float* rm, const float* za,
                       float* __restrict__ sc, float* __restrict__ wsv,
                       float* __restrict__ dpsiT)
{
  const int r = blockIdx.x * blockDim.x + threadIdx.x;
  const float alpha = MIN_ALPHA + (1.0f - MIN_ALPHA) / (1.0f + expf(-za[0]));
  if (r == 0) {
    sc[0] = alpha;
    sc[1] = log1pf(expf(rb[0]));   // beta
    sc[2] = log1pf(expf(rs[0]));   // sigma
    sc[3] = log1pf(expf(rg[0]));   // gamma
    sc[4] = log1pf(expf(rm[0]));   // mu
    sc[5] = powf(DTSTEP, -alpha) / expf(lgammaf(2.0f - alpha));  // C
  }
  if (r < DNM1) {
    float w = powf((float)(r + 1), 1.0f - alpha);
    if (r > 0) w -= powf((float)r, 1.0f - alpha);
    wsv[r] = w;
    #pragma unroll
    for (int c = 0; c < 5; ++c)
      dpsiT[c * DNM1 + r] = y[(r + 1) * 5 + c] - y[r * 5 + c];
  }
}

// ---------------------------------------------------------------------------
// Caputo L1 causal convolution + SEIRD residual + block-partial MSE
// ---------------------------------------------------------------------------
__global__ __launch_bounds__(256) void k_fracres(
    const float* __restrict__ y, const float* __restrict__ wsv,
    const float* __restrict__ dpsiT, const float* __restrict__ sc,
    float* __restrict__ partial)
{
  __shared__ float sacc[8];
  const int lane = threadIdx.x & 31;
  const int wave = threadIdx.x >> 5;
  const int r = blockIdx.x * 8 + wave;
  float d[5] = {0.f, 0.f, 0.f, 0.f, 0.f};
  if (r >= 1) {
    const float* wr = wsv + (r - 1);
    for (int j = lane; j < r; j += 32) {
      const float w = wr[-j];     // ws[r-1-j]
      #pragma unroll
      for (int c = 0; c < 5; ++c) d[c] += w * dpsiT[c * DNM1 + j];
    }
  }
  #pragma unroll
  for (int c = 0; c < 5; ++c)
    #pragma unroll
    for (int o = 16; o > 0; o >>= 1) d[c] += __shfl_xor(d[c], o, 32);

  const float C = sc[5], beta = sc[1], sigma = sc[2], gamma = sc[3], mu = sc[4];
  const float s  = y[r * 5 + 0];
  const float e  = y[r * 5 + 1];
  const float ii = y[r * 5 + 2];
  const float dd = y[r * 5 + 4];
  const float inf_term = beta * s * ii / (1.0f - dd);
  float f[5];
  f[0] = -inf_term;
  f[1] = inf_term - sigma * e;
  f[2] = sigma * e - (gamma + mu) * ii;
  f[3] = gamma * ii;
  f[4] = mu * ii;
  float res = 0.f;
  #pragma unroll
  for (int c = 0; c < 5; ++c) { const float t = C * d[c] - f[c]; res += t * t; }
  if (lane == 0) sacc[wave] = res;
  __syncthreads();
  if (threadIdx.x == 0) {
    float t = 0.f;
    #pragma unroll
    for (int wv = 0; wv < 8; ++wv) t += sacc[wv];
    partial[blockIdx.x] = t;
  }
}

__global__ __launch_bounds__(256) void k_finalize(const float* __restrict__ partial,
                                                  int n, float* __restrict__ out)
{
  __shared__ float s[256];
  float a = 0.f;
  for (int i = threadIdx.x; i < n; i += 256) a += partial[i];
  s[threadIdx.x] = a;
  __syncthreads();
  for (int o = 128; o > 0; o >>= 1) {
    if ((int)threadIdx.x < o) s[threadIdx.x] += s[threadIdx.x + o];
    __syncthreads();
  }
  if (threadIdx.x == 0) out[0] = s[0] / (float)(NPTS * 5);
}

// ---------------------------------------------------------------------------
extern "C" void kernel_launch(void* const* d_in, const int* in_sizes, int n_in,
                              void* d_out, int out_size, void* d_ws, size_t ws_size,
                              hipStream_t stream)
{
  (void)in_sizes; (void)n_in; (void)out_size; (void)ws_size;
  const float* t        = (const float*)d_in[0];
  const float* W_in     = (const float*)d_in[1];
  const float* b_in     = (const float*)d_in[2];
  const float* Wh       = (const float*)d_in[3];
  const float* bh       = (const float*)d_in[4];
  const float* W_out    = (const float*)d_in[5];
  const float* b_out    = (const float*)d_in[6];
  const float* raw_beta  = (const float*)d_in[7];
  const float* raw_sigma = (const float*)d_in[8];
  const float* raw_gamma = (const float*)d_in[9];
  const float* raw_mu    = (const float*)d_in[10];
  const float* z_alpha   = (const float*)d_in[11];
  float* out = (float*)d_out;

  char* ws = (char*)d_ws;
  size_t off = 0;
  auto alloc = [&](size_t bytes) -> void* {
    void* p = ws + off;
    off = (off + bytes + 255) & ~(size_t)255;
    return p;
  };

  float*          Hf    = (float*)alloc((size_t)NPTS * HDIM * 4);
  unsigned short* A0hi  = (unsigned short*)alloc((size_t)NPTS * HDIM * 2);
  unsigned short* A0lo  = (unsigned short*)alloc((size_t)NPTS * HDIM * 2);
  unsigned short* A1hi  = (unsigned short*)alloc((size_t)NPTS * HDIM * 2);
  unsigned short* A1lo  = (unsigned short*)alloc((size_t)NPTS * HDIM * 2);
  unsigned short* Whi   = (unsigned short*)alloc((size_t)NLAYER * HDIM * HDIM * 2);
  unsigned short* Wlo   = (unsigned short*)alloc((size_t)NLAYER * HDIM * HDIM * 2);
  float*          yv    = (float*)alloc((size_t)NPTS * 5 * 4);
  float*          dpsiT = (float*)alloc((size_t)5 * DNM1 * 4);
  float*          wsv   = (float*)alloc((size_t)DNM1 * 4);
  float*          sc    = (float*)alloc(64);
  float*          part  = (float*)alloc((size_t)(NPTS / 8) * 4);

  k_split_w<<<(NLAYER * HDIM * HDIM) / 256, 256, 0, stream>>>(Wh, Whi, Wlo);
  k_input_layer<<<(NPTS * HDIM) / 256, 256, 0, stream>>>(t, W_in, b_in, A0hi, A0lo);

  const unsigned short* inhi = A0hi;
  const unsigned short* inlo = A0lo;
  for (int l = 0; l < NLAYER; ++l) {
    unsigned short* ohi = (l & 1) ? A0hi : A1hi;
    unsigned short* olo = (l & 1) ? A0lo : A1lo;
    k_gemm_tanh<<<dim3(NPTS / BM, HDIM / BN), 256, 0, stream>>>(
        inhi, inlo,
        Whi + (size_t)l * HDIM * HDIM, Wlo + (size_t)l * HDIM * HDIM,
        bh + (size_t)l * HDIM, Hf, ohi, olo);
    inhi = ohi; inlo = olo;
  }

  k_out_softmax<<<NPTS / 8, 256, 0, stream>>>(Hf, W_out, b_out, yv);
  k_prep<<<32, 256, 0, stream>>>(yv, raw_beta, raw_sigma, raw_gamma, raw_mu,
                                 z_alpha, sc, wsv, dpsiT);
  k_fracres<<<NPTS / 8, 256, 0, stream>>>(yv, wsv, dpsiT, sc, part);
  k_finalize<<<1, 256, 0, stream>>>(part, NPTS / 8, out);
}
